// LSTMDecoderWithAttention_26371099197858
// MI455X (gfx1250) — matrix operand order repaired and checked by reading, not compiled, not run on hardware
//
#include <hip/hip_runtime.h>
#include <hip/hip_bf16.h>

// LSTM decoder w/ degenerate attention (softmax over size-1 axis -> attn==1).
// ctx and G0 = ctx@W_ih + b_ih + b_hh are step-invariant; per step only
// gates = G0 + h@W_hh and logits = h@W_lm + b_lm remain. W_lm repacked to
// bf16 WMMA-fragment order (65.5 MB, L2-resident on MI455X's 192 MB L2).
// k_gates: split-K over 8 waves + LDS reduction (serial-latency chain /8).
// k_lmhead: hA staged in LDS per block, 2 N-tiles/wave, B prefetched.

typedef __attribute__((ext_vector_type(16))) __bf16 v16bf;
typedef __attribute__((ext_vector_type(8)))  float  v8f;

#define B_   32
#define N_   196
#define E_   1024
#define D_   1024
#define V_   32000
#define T_   100
#define KT_  (D_ / 32)        // 32 K-chunks of 32
#define NT_G (4 * D_ / 16)    // 256 column tiles of gates matrix
#define NT_LM (V_ / 16)       // 2000 column tiles of lm_head

__device__ __forceinline__ v8f wmma_bf16(v16bf a, v16bf b, v8f c) {
  return __builtin_amdgcn_wmma_f32_16x16x32_bf16(
      /*neg_a=*/false, a, /*neg_b=*/false, b,
      /*c_mod=*/(short)0, c, /*reuse_a=*/false, /*reuse_b=*/false);
}

__device__ __forceinline__ float sigmoid_(float x) {
  return 1.0f / (1.0f + __expf(-x));
}

// F[b,e] = sum_n relu(features[b,n,e])
__global__ void k_rowsum_relu(const float* __restrict__ feat, float* __restrict__ F) {
  int tid = blockIdx.x * blockDim.x + threadIdx.x;
  if (tid >= B_ * E_) return;
  int b = tid / E_, e = tid % E_;
  const float* p = feat + (size_t)b * N_ * E_ + e;
  float s = 0.f;
  for (int n = 0; n < N_; ++n) {
    float v = p[(size_t)n * E_];
    s += (v > 0.f) ? v : 0.f;
  }
  F[tid] = s;
}

// ctx[b,d] = F[b,:] . W_fv[:,d] + N*b_fv[d]
__global__ void k_ctx(const float* __restrict__ F, const float* __restrict__ W_fv,
                      const float* __restrict__ b_fv, float* __restrict__ ctx) {
  int tid = blockIdx.x * blockDim.x + threadIdx.x;
  if (tid >= B_ * D_) return;
  int b = tid / D_, d = tid % D_;
  float s = (float)N_ * b_fv[d];
  const float* f = F + (size_t)b * E_;
  for (int e = 0; e < E_; ++e) s += f[e] * W_fv[(size_t)e * D_ + d];
  ctx[tid] = s;
}

// G0[b,j] = ctx[b,:] . W_ih[:,j] + b_ih[j] + b_hh[j]
__global__ void k_g0(const float* __restrict__ ctx, const float* __restrict__ W_ih,
                     const float* __restrict__ b_ih, const float* __restrict__ b_hh,
                     float* __restrict__ G0) {
  int tid = blockIdx.x * blockDim.x + threadIdx.x;
  if (tid >= B_ * 4 * D_) return;
  int b = tid / (4 * D_), j = tid % (4 * D_);
  float s = b_ih[j] + b_hh[j];
  const float* cr = ctx + (size_t)b * D_;
  for (int d = 0; d < D_; ++d) s += cr[d] * W_ih[(size_t)d * 4 * D_ + j];
  G0[tid] = s;
}

// Pack fp32 [K=1024][ncols] weights into bf16 WMMA B fragments:
// dst[((kt*ntiles + nt)*32 + L)*16 + e] = src[(kt*32 + (L>=16?8:0) + (e>=8?16:0) + (e&7))*ncols + nt*16 + (L&15)]
__global__ void k_pack_b(const float* __restrict__ src, __bf16* __restrict__ dst,
                         int ncols, int ntiles) {
  int tid = blockIdx.x * blockDim.x + threadIdx.x;
  int total = KT_ * ntiles * 32;
  if (tid >= total) return;
  int L = tid & 31;
  int frag = tid >> 5;
  int nt = frag % ntiles;
  int kt = frag / ntiles;
  int n = nt * 16 + (L & 15);
  int kbase = kt * 32 + ((L >= 16) ? 8 : 0);
  __bf16* o = dst + (size_t)tid * 16;
#pragma unroll
  for (int e = 0; e < 16; ++e)
    o[e] = (__bf16)src[(size_t)(kbase + ((e >= 8) ? 16 : 0) + (e & 7)) * ncols + n];
}

// Init: c = 0; pack h0 (= bos_embed broadcast over batch) into A fragments.
// A-frag layout: frag (kt, mt), lane L holds M=L%16, K = kt*32 + (e>=8?16:0)+(L>=16?8:0)+(e&7)
__global__ void k_init(const float* __restrict__ bos, __bf16* __restrict__ hA,
                       float* __restrict__ c) {
  int tid = blockIdx.x * blockDim.x + threadIdx.x;
  if (tid < B_ * D_) c[tid] = 0.f;
  if (tid < KT_ * 2 * 32) {
    int L = tid & 31;
    int kt = (tid >> 5) >> 1;   // mt irrelevant: h0 same for all batch rows
    __bf16* o = hA + (size_t)tid * 16;
#pragma unroll
    for (int e = 0; e < 16; ++e) {
      int k = kt * 32 + ((e >= 8) ? 16 : 0) + ((L >= 16) ? 8 : 0) + (e & 7);
      o[e] = (__bf16)bos[k];
    }
  }
}

// out[0,b,v] = (v==START) ? 1 : 0   (START == 0)
__global__ void k_out0(float* __restrict__ out) {
  int tid = blockIdx.x * blockDim.x + threadIdx.x;
  if (tid >= B_ * V_) return;
  out[tid] = ((tid % V_) == 0) ? 1.0f : 0.0f;
}

// Per step: gates = G0 + h@W_hh. Split-K: 8 waves each do 4 of 32 K-chunks
// with software-pipelined fragment prefetch, partials reduced through LDS,
// then LSTM nonlinearity + repack h_new into next-step A fragments.
__global__ void __launch_bounds__(256)
k_gates(const __bf16* __restrict__ hin, __bf16* __restrict__ hout,
        const __bf16* __restrict__ Whh, const float* __restrict__ G0,
        float* __restrict__ c) {
  __shared__ float red[8][2][4][8][32];   // 64 KB partial C fragments
  const int nt = blockIdx.x;              // 0..63: d-tile within D
  const int lane = threadIdx.x & 31;
  const int w = threadIdx.x >> 5;         // wave id: owns kt = 4w..4w+3

  v8f acc[2][4];
#pragma unroll
  for (int m = 0; m < 2; ++m)
#pragma unroll
    for (int g = 0; g < 4; ++g)
      acc[m][g] = (v8f){0.f, 0.f, 0.f, 0.f, 0.f, 0.f, 0.f, 0.f};

  const size_t astride = (size_t)2 * 32 * 16;      // A elems per K-chunk
  const size_t bstride = (size_t)NT_G * 32 * 16;   // B elems per K-chunk
  const __bf16* pA = hin + ((size_t)(w * 4) * 2 * 32 + lane) * 16;
  const __bf16* pB = Whh + (((size_t)(w * 4) * NT_G + nt) * 32 + lane) * 16;

  v16bf a0 = *(const v16bf*)(pA);
  v16bf a1 = *(const v16bf*)(pA + 32 * 16);
  v16bf bf[4];
#pragma unroll
  for (int g = 0; g < 4; ++g)
    bf[g] = *(const v16bf*)(pB + (size_t)g * (64 * 32 * 16));

#pragma unroll
  for (int i = 0; i < 4; ++i) {
    int nx = (i + 1 < 4) ? (i + 1) : i;  // prefetch next (clamped, uniform)
    v16bf na0 = *(const v16bf*)(pA + (size_t)nx * astride);
    v16bf na1 = *(const v16bf*)(pA + (size_t)nx * astride + 32 * 16);
    v16bf nb[4];
#pragma unroll
    for (int g = 0; g < 4; ++g)
      nb[g] = *(const v16bf*)(pB + (size_t)nx * bstride + (size_t)g * (64 * 32 * 16));
#pragma unroll
    for (int g = 0; g < 4; ++g) {
      acc[0][g] = wmma_bf16(a0, bf[g], acc[0][g]);
      acc[1][g] = wmma_bf16(a1, bf[g], acc[1][g]);
    }
    a0 = na0; a1 = na1;
#pragma unroll
    for (int g = 0; g < 4; ++g) bf[g] = nb[g];
  }

#pragma unroll
  for (int m = 0; m < 2; ++m)
#pragma unroll
    for (int g = 0; g < 4; ++g)
#pragma unroll
      for (int v = 0; v < 8; ++v)
        red[w][m][g][v][lane] = acc[m][g][v];
  __syncthreads();

  // 512 (b,d) elements per block; each thread finalizes 2.
#pragma unroll
  for (int rep = 0; rep < 2; ++rep) {
    int idx = threadIdx.x + rep * 256;
    int mt = idx >> 8;
    int v = (idx >> 5) & 7;
    int ln = idx & 31;
    float s0 = 0.f, s1 = 0.f, s2 = 0.f, s3 = 0.f;
#pragma unroll
    for (int ww = 0; ww < 8; ++ww) {
      s0 += red[ww][mt][0][v][ln];
      s1 += red[ww][mt][1][v][ln];
      s2 += red[ww][mt][2][v][ln];
      s3 += red[ww][mt][3][v][ln];
    }
    int b = mt * 16 + ((ln >= 16) ? 8 : 0) + v;  // C layout: M = v + (L>=16?8:0)
    int d = nt * 16 + (ln & 15);
    float gi = s0 + G0[(size_t)b * 4 * D_ + 0 * D_ + d];
    float gf = s1 + G0[(size_t)b * 4 * D_ + 1 * D_ + d];
    float gg = s2 + G0[(size_t)b * 4 * D_ + 2 * D_ + d];
    float go = s3 + G0[(size_t)b * 4 * D_ + 3 * D_ + d];
    float cold = c[(size_t)b * D_ + d];
    float cn = sigmoid_(gf) * cold + sigmoid_(gi) * tanhf(gg);
    float hn = sigmoid_(go) * tanhf(cn);
    c[(size_t)b * D_ + d] = cn;
    // repack into next-step A fragment position
    int kth = d >> 5;
    int koff = d & 31;
    int Lp = (b & 15) + (((koff & 15) >= 8) ? 16 : 0);
    int ep = ((koff >= 16) ? 8 : 0) + (koff & 7);
    int mtp = b >> 4;
    hout[((size_t)((kth * 2 + mtp) * 32 + Lp)) * 16 + ep] = (__bf16)hn;
  }
}

// logits = h @ W_lm + b_lm. hA staged once per block into LDS (cuts global
// A traffic 16x), each wave computes two 16-col tiles x both 16-row batch
// tiles (4 WMMAs / K-chunk), B fragments prefetched one K-chunk ahead.
__global__ void __launch_bounds__(256)
k_lmhead(const __bf16* __restrict__ hA, const __bf16* __restrict__ Wlm,
         const float* __restrict__ b_lm, float* __restrict__ outp) {
  __shared__ __bf16 sha[KT_ * 2 * 32 * 16];   // 64 KB packed h fragments
  {
    const uint4* src = (const uint4*)hA;
    uint4* dst = (uint4*)sha;
#pragma unroll
    for (int i = 0; i < 8; ++i)
      dst[threadIdx.x + 256 * i] = src[threadIdx.x + 256 * i];
  }
  __syncthreads();

  const int lane = threadIdx.x & 31;
  const int wave = threadIdx.x >> 5;
  const int nt0 = blockIdx.x * 16 + wave * 2;  // 125 blocks * 16 tiles = 2000
  const int nt1 = nt0 + 1;

  v8f acc00 = (v8f){0.f, 0.f, 0.f, 0.f, 0.f, 0.f, 0.f, 0.f};
  v8f acc10 = acc00, acc01 = acc00, acc11 = acc00;

  const size_t bstride = (size_t)NT_LM * 32 * 16;  // B elems per K-chunk
  const __bf16* pB0 = Wlm + ((size_t)nt0 * 32 + lane) * 16;
  const __bf16* pB1 = Wlm + ((size_t)nt1 * 32 + lane) * 16;

  v16bf b0 = *(const v16bf*)pB0;
  v16bf b1 = *(const v16bf*)pB1;
  for (int kt = 0; kt < KT_; ++kt) {
    int nx = (kt + 1 < KT_) ? (kt + 1) : kt;   // uniform clamped prefetch
    v16bf nb0 = *(const v16bf*)(pB0 + (size_t)nx * bstride);
    v16bf nb1 = *(const v16bf*)(pB1 + (size_t)nx * bstride);
    v16bf a0 = *(const v16bf*)(sha + ((size_t)((kt * 2 + 0) * 32 + lane)) * 16);
    v16bf a1 = *(const v16bf*)(sha + ((size_t)((kt * 2 + 1) * 32 + lane)) * 16);
    acc00 = wmma_bf16(a0, b0, acc00);
    acc10 = wmma_bf16(a1, b0, acc10);
    acc01 = wmma_bf16(a0, b1, acc01);
    acc11 = wmma_bf16(a1, b1, acc11);
    b0 = nb0; b1 = nb1;
  }

  const int hi = lane >> 4;
  const int col = lane & 15;
  const int n0 = nt0 * 16 + col;
  const int n1 = nt1 * 16 + col;
  const float bias0 = b_lm[n0];
  const float bias1 = b_lm[n1];
#pragma unroll
  for (int v = 0; v < 8; ++v) {
    int br0 = hi * 8 + v;
    int br1 = 16 + hi * 8 + v;
    outp[(size_t)br0 * V_ + n0] = acc00[v] + bias0;
    outp[(size_t)br1 * V_ + n0] = acc10[v] + bias0;
    outp[(size_t)br0 * V_ + n1] = acc01[v] + bias1;
    outp[(size_t)br1 * V_ + n1] = acc11[v] + bias1;
  }
}

extern "C" void kernel_launch(void* const* d_in, const int* in_sizes, int n_in,
                              void* d_out, int out_size, void* d_ws, size_t ws_size,
                              hipStream_t stream) {
  const float* features = (const float*)d_in[0];
  // d_in[1]=W_fk, d_in[2]=b_fk, d_in[5]=W_tk, d_in[6]=b_tk: dead (attn==1)
  const float* W_fv = (const float*)d_in[3];
  const float* b_fv = (const float*)d_in[4];
  const float* W_ih = (const float*)d_in[7];
  const float* W_hh = (const float*)d_in[8];
  const float* b_ih = (const float*)d_in[9];
  const float* b_hh = (const float*)d_in[10];
  const float* W_lm = (const float*)d_in[11];
  const float* b_lm = (const float*)d_in[12];
  const float* bos  = (const float*)d_in[13];
  float* out = (float*)d_out;

  // Workspace carve-up (~74.9 MB total)
  char* w = (char*)d_ws;
  __bf16* Wlm_p = (__bf16*)w; w += (size_t)KT_ * NT_LM * 32 * 16 * 2;  // 65,536,000
  __bf16* Whh_p = (__bf16*)w; w += (size_t)KT_ * NT_G  * 32 * 16 * 2;  //  8,388,608
  float* Fsum = (float*)w;  w += (size_t)B_ * E_ * 4;
  float* ctx  = (float*)w;  w += (size_t)B_ * D_ * 4;
  float* G0   = (float*)w;  w += (size_t)B_ * 4 * D_ * 4;
  float* cst  = (float*)w;  w += (size_t)B_ * D_ * 4;
  __bf16* hA0 = (__bf16*)w; w += (size_t)KT_ * 2 * 32 * 16 * 2;        // 65,536
  __bf16* hA1 = (__bf16*)w;

  // One-time precompute
  k_pack_b<<<(KT_ * NT_LM * 32 + 255) / 256, 256, 0, stream>>>(W_lm, Wlm_p, V_, NT_LM);
  k_pack_b<<<(KT_ * NT_G * 32 + 255) / 256, 256, 0, stream>>>(W_hh, Whh_p, 4 * D_, NT_G);
  k_rowsum_relu<<<(B_ * E_ + 255) / 256, 256, 0, stream>>>(features, Fsum);
  k_ctx<<<(B_ * D_ + 255) / 256, 256, 0, stream>>>(Fsum, W_fv, b_fv, ctx);
  k_g0<<<(B_ * 4 * D_ + 255) / 256, 256, 0, stream>>>(ctx, W_ih, b_ih, b_hh, G0);
  k_init<<<(B_ * D_ + 255) / 256, 256, 0, stream>>>(bos, hA0, cst);
  k_out0<<<(B_ * V_ + 255) / 256, 256, 0, stream>>>(out);

  // 99 sequential decode steps; double-buffered packed h
  for (int t = 1; t < T_; ++t) {
    __bf16* hin  = (t & 1) ? hA0 : hA1;
    __bf16* hout = (t & 1) ? hA1 : hA0;
    k_gates<<<64, 256, 0, stream>>>(hin, hout, Whh_p, G0, cst);
    k_lmhead<<<125, 256, 0, stream>>>(hout, Wlm_p, b_lm, out + (size_t)t * B_ * V_);
  }
}
